// MultiViewGraphAttentionNetWork_31653908972241
// MI455X (gfx1250) — 16-bit and 8-bit matrix operand order repaired and checked by reading, compiled, not run on hardware
//
#include <hip/hip_runtime.h>
#include <hip/hip_bf16.h>

typedef __attribute__((ext_vector_type(16))) _Float16 v16h;
typedef __attribute__((ext_vector_type(8)))  float    v8f;
typedef __attribute__((ext_vector_type(8)))  int      v8i;

#define U_N 6000
#define V_N 4000
#define S_N 400
#define E_N 64
#define U_P 6016
#define V_P 4032
#define S_P 448

__global__ void k_pack_f16(const float* __restrict__ src, _Float16* __restrict__ dst,
                           int R, int C, int ldd, int transpose) {
  long long i = (long long)blockIdx.x * blockDim.x + threadIdx.x;
  if (!transpose) {
    long long tot = (long long)R * ldd;
    if (i >= tot) return;
    int r = (int)(i / ldd), c = (int)(i % ldd);
    dst[i] = (c < C) ? (_Float16)src[(long long)r * C + c] : (_Float16)0.0f;
  } else {
    long long tot = (long long)C * ldd;
    if (i >= tot) return;
    int r = (int)(i / ldd), c = (int)(i % ldd);
    dst[i] = (c < R) ? (_Float16)src[(long long)c * C + r] : (_Float16)0.0f;
  }
}

__global__ void k_pack_u8(const float* __restrict__ src, unsigned char* __restrict__ dst,
                          int R, int C, int ldd, int transpose) {
  long long i = (long long)blockIdx.x * blockDim.x + threadIdx.x;
  if (!transpose) {
    long long tot = (long long)R * ldd;
    if (i >= tot) return;
    int r = (int)(i / ldd), c = (int)(i % ldd);
    dst[i] = (c < C && src[(long long)r * C + c] != 0.0f) ? 1 : 0;
  } else {
    long long tot = (long long)C * ldd;
    if (i >= tot) return;
    int r = (int)(i / ldd), c = (int)(i % ldd);
    dst[i] = (c < R && src[(long long)c * C + r] != 0.0f) ? 1 : 0;
  }
}

__global__ void k_u8f16(const unsigned char* __restrict__ s, _Float16* __restrict__ d, long long n) {
  long long i = (long long)blockIdx.x * blockDim.x + threadIdx.x;
  if (i < n) d[i] = (_Float16)(float)s[i];
}

__global__ void k_rowsum_f32(const float* __restrict__ M, float* __restrict__ out, int R, int C) {
  int r = blockIdx.x * blockDim.x + threadIdx.x;
  if (r >= R) return;
  float s = 0.f;
  for (int c = 0; c < C; ++c) s += M[(long long)r * C + c];
  out[r] = s;
}

__global__ void k_vecmat_col(const float* __restrict__ M, const float* __restrict__ x,
                             float* __restrict__ out, int R, int C) {
  int c = blockIdx.x * blockDim.x + threadIdx.x;
  if (c >= C) return;
  float s = 0.f;
  for (int r = 0; r < R; ++r) s += (x ? x[r] : 1.0f) * M[(long long)r * C + c];
  out[c] = s;
}

__global__ void k_matvec_row(const float* __restrict__ M, const float* __restrict__ x,
                             float* __restrict__ out, int R, int C) {
  int r = blockIdx.x * blockDim.x + threadIdx.x;
  if (r >= R) return;
  float s = 0.f;
  for (int c = 0; c < C; ++c) s += M[(long long)r * C + c] * x[c];
  out[r] = s;
}

__global__ void k_recip_clamp(const float* __restrict__ s, float* __restrict__ d, int n) {
  int i = blockIdx.x * blockDim.x + threadIdx.x;
  if (i < n) d[i] = 1.0f / fmaxf(1.0f, s[i]);
}

__global__ void k_rowsum_recip_u8(const unsigned char* __restrict__ m, float* __restrict__ out,
                                  int R, int ld) {
  int r = blockIdx.x * blockDim.x + threadIdx.x;
  if (r >= R) return;
  int s = 0;
  for (int c = 0; c < ld; ++c) s += m[(long long)r * ld + c];
  out[r] = 1.0f / fmaxf(1.0f, (float)s);
}

__global__ void k_count_recip(const float* __restrict__ r, int n, float* __restrict__ out) {
  __shared__ int sd[256];
  int cnt = 0;
  for (int i = threadIdx.x; i < n; i += 256) cnt += (r[i] > 0.f) ? 1 : 0;
  sd[threadIdx.x] = cnt;
  __syncthreads();
  for (int off = 128; off > 0; off >>= 1) {
    if ((int)threadIdx.x < off) sd[threadIdx.x] += sd[threadIdx.x + off];
    __syncthreads();
  }
  if (threadIdx.x == 0) out[0] = 1.0f / fmaxf(1.0f, (float)sd[0]);
}

__global__ void k_bcast_mask(const float* __restrict__ r, _Float16* __restrict__ mask,
                             int C, long long total, int ldd) {
  long long i = (long long)blockIdx.x * blockDim.x + threadIdx.x;
  if (i >= total) return;
  int c = (int)(i % ldd);
  mask[i] = (c < C && r[c] > 0.f) ? (_Float16)1.0f : (_Float16)0.0f;
}

__global__ void k_fill(float* __restrict__ d, const float* __restrict__ s, int n) {
  int i = blockIdx.x * blockDim.x + threadIdx.x;
  if (i < n) d[i] = s[0];
}

__global__ void k_zero(float* __restrict__ d, int n) {
  int i = blockIdx.x * blockDim.x + threadIdx.x;
  if (i < n) d[i] = 0.f;
}

__global__ __launch_bounds__(128)
void k_gemm_f16(const _Float16* __restrict__ A, const _Float16* __restrict__ Bt,
                float* __restrict__ C, int M, int N, int K, int ldA, int ldC,
                const float* __restrict__ rowscale, const float* __restrict__ bias, int act) {
  const int lane = threadIdx.x & 31;
  const int w    = threadIdx.x >> 5;
  const int m0 = blockIdx.y * 64 + w * 16;
  const int n0 = blockIdx.x * 32;
  const int arow  = m0 + (lane & 15);
  const int bcol0 = n0 + (lane & 15);
  const int bcol1 = bcol0 + 16;
  const int kh  = (lane >> 4) * 8;
  const int kbB = (lane >> 4) * 8;
  const bool rok = arow < M;
  const bool c0ok = bcol0 < N;
  const bool c1ok = bcol1 < N;
  v8f acc0 = {}, acc1 = {};
  for (int k0 = 0; k0 < K; k0 += 32) {
    v16h a = {}, b0 = {}, b1 = {};
    if (rok) {
      const _Float16* ap = A + (long long)arow * ldA + k0 + kh;
      __builtin_prefetch(ap + 32, 0, 1);
      #pragma unroll
      for (int v = 0; v < 8; ++v) {
        int kb = ((v >> 2) << 4) + ((v & 3) << 1);
        a[2 * v]     = ap[kb];
        a[2 * v + 1] = ap[kb + 1];
      }
    }
    if (c0ok) {
      const _Float16* bp = Bt + (long long)bcol0 * ldA + k0 + kbB;
      __builtin_prefetch(bp + 32, 0, 1);
      #pragma unroll
      for (int v = 0; v < 8; ++v) {
        int kb = ((v >> 2) << 4) + ((v & 3) << 1);
        b0[2 * v]     = bp[kb];
        b0[2 * v + 1] = bp[kb + 1];
      }
    }
    if (c1ok) {
      const _Float16* bp = Bt + (long long)bcol1 * ldA + k0 + kbB;
      #pragma unroll
      for (int v = 0; v < 8; ++v) {
        int kb = ((v >> 2) << 4) + ((v & 3) << 1);
        b1[2 * v]     = bp[kb];
        b1[2 * v + 1] = bp[kb + 1];
      }
    }
    acc0 = __builtin_amdgcn_wmma_f32_16x16x32_f16(false, a, false, b0, (short)0, acc0, false, false);
    acc1 = __builtin_amdgcn_wmma_f32_16x16x32_f16(false, a, false, b1, (short)0, acc1, false, false);
  }
  #pragma unroll
  for (int t = 0; t < 2; ++t) {
    int scol = n0 + t * 16 + (lane & 15);
    if (scol >= N) continue;
    #pragma unroll
    for (int r = 0; r < 8; ++r) {
      int srow = m0 + r + kh;
      if (srow >= M) continue;
      float v = (t == 0) ? acc0[r] : acc1[r];
      if (rowscale) v *= rowscale[srow];
      if (bias)     v += bias[scol];
      if (act == 1) v = (v > 0.f) ? v : 0.2f * v;
      C[(long long)srow * ldC + scol] = v;
    }
  }
}

__global__ __launch_bounds__(128)
void k_gemm_bin(const unsigned char* __restrict__ A, const unsigned char* __restrict__ Bt,
                unsigned char* __restrict__ D, int M, int N, int K, int ldA, int ldD,
                int zero_diag) {
  const int lane = threadIdx.x & 31;
  const int w    = threadIdx.x >> 5;
  const int m0 = blockIdx.y * 64 + (w >> 1) * 32;
  const int n0 = blockIdx.x * 64 + (w & 1) * 32;
  const int arow0 = m0 + (lane & 15);
  const int arow1 = arow0 + 16;
  const int bcol0 = n0 + (lane & 15);
  const int bcol1 = bcol0 + 16;
  const int kh  = (lane >> 4) * 8;
  const int kbB = (lane >> 4) * 8;
  const bool r0ok = arow0 < M, r1ok = arow1 < M;
  const bool c0ok = bcol0 < N, c1ok = bcol1 < N;
  v8i acc00 = {}, acc01 = {}, acc10 = {}, acc11 = {};
  for (int k0 = 0; k0 < K; k0 += 64) {
    v8i a0 = {}, a1 = {}, b0 = {}, b1 = {};
    if (r0ok) {
      const unsigned char* ap = A + (long long)arow0 * ldA + k0 + kh;
      __builtin_prefetch(ap + 64, 0, 1);
      #pragma unroll
      for (int v = 0; v < 8; ++v) {
        int kb = ((v >> 1) << 4) + ((v & 1) << 2);
        a0[v] = *(const int*)(ap + kb);
      }
    }
    if (r1ok) {
      const unsigned char* ap = A + (long long)arow1 * ldA + k0 + kh;
      #pragma unroll
      for (int v = 0; v < 8; ++v) {
        int kb = ((v >> 1) << 4) + ((v & 1) << 2);
        a1[v] = *(const int*)(ap + kb);
      }
    }
    if (c0ok) {
      const unsigned char* bp = Bt + (long long)bcol0 * ldA + k0 + kbB;
      __builtin_prefetch(bp + 64, 0, 1);
      #pragma unroll
      for (int v = 0; v < 8; ++v) {
        int kb = ((v >> 1) << 4) + ((v & 1) << 2);
        b0[v] = *(const int*)(bp + kb);
      }
    }
    if (c1ok) {
      const unsigned char* bp = Bt + (long long)bcol1 * ldA + k0 + kbB;
      #pragma unroll
      for (int v = 0; v < 8; ++v) {
        int kb = ((v >> 1) << 4) + ((v & 1) << 2);
        b1[v] = *(const int*)(bp + kb);
      }
    }
    acc00 = __builtin_amdgcn_wmma_i32_16x16x64_iu8(false, a0, false, b0, acc00, false, false);
    acc01 = __builtin_amdgcn_wmma_i32_16x16x64_iu8(false, a0, false, b1, acc01, false, false);
    acc10 = __builtin_amdgcn_wmma_i32_16x16x64_iu8(false, a1, false, b0, acc10, false, false);
    acc11 = __builtin_amdgcn_wmma_i32_16x16x64_iu8(false, a1, false, b1, acc11, false, false);
  }
  #pragma unroll
  for (int tm = 0; tm < 2; ++tm) {
    #pragma unroll
    for (int tn = 0; tn < 2; ++tn) {
      int scol = n0 + tn * 16 + (lane & 15);
      if (scol >= ldD) continue;
      #pragma unroll
      for (int r = 0; r < 8; ++r) {
        int srow = m0 + tm * 16 + r + kh;
        if (srow >= M) continue;
        int av = (tm == 0) ? ((tn == 0) ? acc00[r] : acc01[r])
                           : ((tn == 0) ? acc10[r] : acc11[r]);
        unsigned char o = 0;
        if (scol < N && av > 0 && !(zero_diag && srow == scol)) o = 1;
        D[(long long)srow * ldD + scol] = o;
      }
    }
  }
}

__global__ void k_attr_p(const float* __restrict__ h, const float* __restrict__ Wp,
                         const float* __restrict__ Bp, float* __restrict__ p, int R) {
  long long i = (long long)blockIdx.x * blockDim.x + threadIdx.x;
  if (i >= (long long)R * 3) return;
  int row = (int)(i / 3), j = (int)(i % 3);
  float s = Bp[j];
  for (int e = 0; e < E_N; ++e) s += h[(long long)row * E_N + e] * Wp[e * 3 + j];
  p[i] = tanhf(s);
}

__global__ void k_attr_H(const float* __restrict__ p, const float* __restrict__ X,
                         const float* __restrict__ Vm, float* __restrict__ H, int R) {
  long long i = (long long)blockIdx.x * blockDim.x + threadIdx.x;
  if (i >= (long long)R * E_N) return;
  int row = (int)(i / E_N), e = (int)(i % E_N);
  float s = 0.f;
  #pragma unroll
  for (int j = 0; j < 3; ++j) s += p[(long long)row * 3 + j] * X[j] * Vm[e * 3 + j];
  H[i] = s;
}

__global__ void k_attr_score(const float* __restrict__ H, const float* __restrict__ Wq,
                             const float* __restrict__ Bq, const float* __restrict__ Q,
                             int R, float* __restrict__ eacc) {
  __shared__ float sd[256];
  int i = blockIdx.x * blockDim.x + threadIdx.x;
  float t = 0.f;
  if (i < R) {
    #pragma unroll
    for (int d = 0; d < 4; ++d) {
      float s = Bq[d];
      for (int e = 0; e < E_N; ++e) s += H[(long long)i * E_N + e] * Wq[e * 4 + d];
      t += tanhf(s) * Q[d];
    }
  }
  sd[threadIdx.x] = t;
  __syncthreads();
  for (int off = 128; off > 0; off >>= 1) {
    if ((int)threadIdx.x < off) sd[threadIdx.x] += sd[threadIdx.x + off];
    __syncthreads();
  }
  if (threadIdx.x == 0) atomicAdd(eacc, sd[0]);
}

__global__ void k_attr_combine(const float* __restrict__ H, const float* __restrict__ esc,
                               int Mcnt, int R, float* __restrict__ out) {
  long long i = (long long)blockIdx.x * blockDim.x + threadIdx.x;
  long long tot = (long long)R * E_N;
  if (i >= tot) return;
  float b[3], mx = -1e30f;
  for (int m = 0; m < Mcnt; ++m) { b[m] = esc[m] / (float)R; mx = fmaxf(mx, b[m]); }
  float ss = 0.f;
  for (int m = 0; m < Mcnt; ++m) { b[m] = expf(b[m] - mx); ss += b[m]; }
  float acc = 0.f;
  for (int m = 0; m < Mcnt; ++m) acc += (b[m] / ss) * H[(long long)m * tot + i];
  out[i] = acc;
}

__global__ void k_fuse4(const float* __restrict__ a, const float* __restrict__ b,
                        const float* __restrict__ c, const float* __restrict__ d,
                        const float* __restrict__ w, long long n, float* __restrict__ out) {
  long long i = (long long)blockIdx.x * blockDim.x + threadIdx.x;
  if (i >= n) return;
  float w0 = w[0], w1 = w[1], w2 = w[2], w3 = w[3];
  float mx = fmaxf(fmaxf(w0, w1), fmaxf(w2, w3));
  float e0 = expf(w0 - mx), e1 = expf(w1 - mx), e2 = expf(w2 - mx), e3 = expf(w3 - mx);
  float s = e0 + e1 + e2 + e3;
  out[i] = (e0 * a[i] + e1 * b[i] + e2 * c[i] + e3 * d[i]) / s;
}

extern "C" void kernel_launch(void* const* d_in, const int* in_sizes, int n_in,
                              void* d_out, int out_size, void* d_ws, size_t ws_size,
                              hipStream_t stream) {
  (void)in_sizes; (void)n_in; (void)out_size; (void)ws_size;
  const float* users    = (const float*)d_in[0];
  const float* products = (const float*)d_in[1];
  const float* Af       = (const float*)d_in[2];
  const float* ua0      = (const float*)d_in[3];
  const float* ua1      = (const float*)d_in[4];
  const float* pa0      = (const float*)d_in[5];
  const float* pa1      = (const float*)d_in[6];
  const float* pa2      = (const float*)d_in[7];
  const float* weight   = (const float*)d_in[8];
  const float* ubias    = (const float*)d_in[9];
  const float* pbias    = (const float*)d_in[10];
  const float* uX  = (const float*)d_in[11];
  const float* uVv = (const float*)d_in[12];
  const float* uWp = (const float*)d_in[13];
  const float* uBp = (const float*)d_in[14];
  const float* uWq = (const float*)d_in[15];
  const float* uBq = (const float*)d_in[16];
  const float* uQ  = (const float*)d_in[17];
  const float* pX  = (const float*)d_in[18];
  const float* pVv = (const float*)d_in[19];
  const float* pWp = (const float*)d_in[20];
  const float* pBp = (const float*)d_in[21];
  const float* pWq = (const float*)d_in[22];
  const float* pBq = (const float*)d_in[23];
  const float* pQ  = (const float*)d_in[24];
  const float* uvw = (const float*)d_in[25];
  const float* pvw = (const float*)d_in[26];
  float* out_u = (float*)d_out;
  float* out_v = (float*)d_out + (size_t)U_N * E_N;

  char* wp = (char*)d_ws;
  auto alloc = [&](size_t bytes) -> char* {
    char* p = wp; wp += (bytes + 255) & ~(size_t)255; return p;
  };
  _Float16* A16     = (_Float16*)alloc((size_t)U_N * V_P * 2);
  _Float16* At16    = (_Float16*)alloc((size_t)V_N * U_P * 2);
  _Float16* prodT16 = (_Float16*)alloc((size_t)E_N * V_P * 2);
  _Float16* userT16 = (_Float16*)alloc((size_t)E_N * U_P * 2);
  _Float16* W16T    = (_Float16*)alloc((size_t)E_N * E_N * 2);
  _Float16* t16u    = (_Float16*)alloc((size_t)U_N * E_N * 2);
  _Float16* t16v    = (_Float16*)alloc((size_t)V_N * E_N * 2);
  _Float16* uT16    = (_Float16*)alloc((size_t)E_N * U_P * 2);
  _Float16* vT16    = (_Float16*)alloc((size_t)E_N * V_P * 2);
  _Float16* mask16  = (_Float16*)alloc((size_t)U_N * V_P * 2);
  unsigned char* ua0_8 = (unsigned char*)alloc((size_t)U_N * S_P);
  unsigned char* ua1_8 = (unsigned char*)alloc((size_t)U_N * S_P);
  unsigned char* pa0_8 = (unsigned char*)alloc((size_t)V_N * S_P);
  unsigned char* pa1_8 = (unsigned char*)alloc((size_t)V_N * S_P);
  unsigned char* pa2_8 = (unsigned char*)alloc((size_t)V_N * S_P);
  unsigned char* A8    = (unsigned char*)alloc((size_t)U_N * V_P);
  unsigned char* At8   = (unsigned char*)alloc((size_t)V_N * U_P);
  unsigned char* sq8   = (unsigned char*)alloc((size_t)U_N * U_P);
  unsigned char* mask8 = (unsigned char*)alloc((size_t)U_N * V_P);
  float* du_raw = (float*)alloc(U_N * 4);
  float* dv_raw = (float*)alloc(V_N * 4);
  float* du_r   = (float*)alloc(U_N * 4);
  float* dv_r   = (float*)alloc(V_N * 4);
  float* vecS   = (float*)alloc(S_P * 4);
  float* vecN   = (float*)alloc(U_N * 4);
  float* scal   = (float*)alloc(8 * 4);
  float* rowsc  = (float*)alloc(U_N * 4);
  float* t32u   = (float*)alloc((size_t)U_N * E_N * 4);
  float* t32v   = (float*)alloc((size_t)V_N * E_N * 4);
  float* u1     = (float*)alloc((size_t)U_N * E_N * 4);
  float* v1     = (float*)alloc((size_t)V_N * E_N * 4);
  float* view1_u = (float*)alloc((size_t)U_N * E_N * 4);
  float* view2_u = (float*)alloc((size_t)U_N * E_N * 4);
  float* view3_u = (float*)alloc((size_t)U_N * E_N * 4);
  float* view4_u = (float*)alloc((size_t)U_N * E_N * 4);
  float* view1_v = (float*)alloc((size_t)V_N * E_N * 4);
  float* view2_v = (float*)alloc((size_t)V_N * E_N * 4);
  float* view3_v = (float*)alloc((size_t)V_N * E_N * 4);
  float* view4_v = (float*)alloc((size_t)V_N * E_N * 4);
  float* h32 = (float*)alloc((size_t)U_N * E_N * 4);
  float* p32 = (float*)alloc((size_t)U_N * 4 * 4);
  float* H3  = (float*)alloc((size_t)3 * U_N * E_N * 4);
  float* esc = (float*)alloc(8 * 4);

  auto cdiv = [](long long a, long long b) -> unsigned { return (unsigned)((a + b - 1) / b); };
  auto g1 = [&](long long n) { return dim3(cdiv(n, 256)); };

  k_pack_f16<<<g1((long long)U_N * V_P), 256, 0, stream>>>(Af, A16, U_N, V_N, V_P, 0);
  k_pack_f16<<<g1((long long)V_N * U_P), 256, 0, stream>>>(Af, At16, U_N, V_N, U_P, 1);
  k_pack_f16<<<g1((long long)E_N * V_P), 256, 0, stream>>>(products, prodT16, V_N, E_N, V_P, 1);
  k_pack_f16<<<g1((long long)E_N * U_P), 256, 0, stream>>>(users, userT16, U_N, E_N, U_P, 1);
  k_pack_f16<<<g1((long long)E_N * E_N), 256, 0, stream>>>(weight, W16T, E_N, E_N, E_N, 1);
  k_pack_u8<<<g1((long long)U_N * S_P), 256, 0, stream>>>(ua0, ua0_8, U_N, S_N, S_P, 0);
  k_pack_u8<<<g1((long long)U_N * S_P), 256, 0, stream>>>(ua1, ua1_8, U_N, S_N, S_P, 0);
  k_pack_u8<<<g1((long long)V_N * S_P), 256, 0, stream>>>(pa0, pa0_8, V_N, S_N, S_P, 0);
  k_pack_u8<<<g1((long long)V_N * S_P), 256, 0, stream>>>(pa1, pa1_8, V_N, S_N, S_P, 0);
  k_pack_u8<<<g1((long long)V_N * S_P), 256, 0, stream>>>(pa2, pa2_8, V_N, S_N, S_P, 0);
  k_pack_u8<<<g1((long long)U_N * V_P), 256, 0, stream>>>(Af, A8, U_N, V_N, V_P, 0);
  k_pack_u8<<<g1((long long)V_N * U_P), 256, 0, stream>>>(Af, At8, U_N, V_N, U_P, 1);

  k_rowsum_f32<<<g1(U_N), 256, 0, stream>>>(Af, du_raw, U_N, V_N);
  k_vecmat_col<<<g1(V_N), 256, 0, stream>>>(Af, nullptr, dv_raw, U_N, V_N);
  k_recip_clamp<<<g1(U_N), 256, 0, stream>>>(du_raw, du_r, U_N);
  k_recip_clamp<<<g1(V_N), 256, 0, stream>>>(dv_raw, dv_r, V_N);

  dim3 gU(cdiv(E_N, 32), cdiv(U_N, 64)), gV(cdiv(E_N, 32), cdiv(V_N, 64));
  k_gemm_f16<<<gU, 128, 0, stream>>>(A16, prodT16, t32u, U_N, E_N, V_P, V_P, E_N, du_r, nullptr, 0);
  k_pack_f16<<<g1((long long)U_N * E_N), 256, 0, stream>>>(t32u, t16u, U_N, E_N, E_N, 0);
  k_gemm_f16<<<gU, 128, 0, stream>>>(t16u, W16T, u1, U_N, E_N, E_N, E_N, E_N, nullptr, ubias, 1);
  k_gemm_f16<<<gV, 128, 0, stream>>>(At16, userT16, t32v, V_N, E_N, U_P, U_P, E_N, dv_r, nullptr, 0);
  k_pack_f16<<<g1((long long)V_N * E_N), 256, 0, stream>>>(t32v, t16v, V_N, E_N, E_N, 0);
  k_gemm_f16<<<gV, 128, 0, stream>>>(t16v, W16T, v1, V_N, E_N, E_N, E_N, E_N, nullptr, pbias, 1);
  k_pack_f16<<<g1((long long)E_N * V_P), 256, 0, stream>>>(v1, vT16, V_N, E_N, V_P, 1);
  k_pack_f16<<<g1((long long)E_N * U_P), 256, 0, stream>>>(u1, uT16, U_N, E_N, U_P, 1);
  k_gemm_f16<<<gU, 128, 0, stream>>>(A16, vT16, t32u, U_N, E_N, V_P, V_P, E_N, du_r, nullptr, 0);
  k_pack_f16<<<g1((long long)U_N * E_N), 256, 0, stream>>>(t32u, t16u, U_N, E_N, E_N, 0);
  k_gemm_f16<<<gU, 128, 0, stream>>>(t16u, W16T, view1_u, U_N, E_N, E_N, E_N, E_N, nullptr, ubias, 1);
  k_gemm_f16<<<gV, 128, 0, stream>>>(At16, uT16, t32v, V_N, E_N, U_P, U_P, E_N, dv_r, nullptr, 0);
  k_pack_f16<<<g1((long long)V_N * E_N), 256, 0, stream>>>(t32v, t16v, V_N, E_N, E_N, 0);
  k_gemm_f16<<<gV, 128, 0, stream>>>(t16v, W16T, view1_v, V_N, E_N, E_N, E_N, E_N, nullptr, pbias, 1);

  auto mask_rank1 = [&](int R, int Cdim, int Kp) {
    k_count_recip<<<1, 256, 0, stream>>>(vecN, Cdim, scal);
    k_bcast_mask<<<g1((long long)R * Kp), 256, 0, stream>>>(vecN, mask16, Cdim, (long long)R * Kp, Kp);
    k_fill<<<g1(R), 256, 0, stream>>>(rowsc, scal, R);
  };
  auto mask_mp3 = [&](const unsigned char* m0_8, int R, int Rp,
                      const unsigned char* m2Bt, int Cdim, int Kp) {
    dim3 gs(cdiv(Rp, 64), cdiv(R, 64));
    k_gemm_bin<<<gs, 128, 0, stream>>>(m0_8, m0_8, sq8, R, R, S_P, S_P, Rp, 1);
    dim3 gm(cdiv(Kp, 64), cdiv(R, 64));
    k_gemm_bin<<<gm, 128, 0, stream>>>(sq8, m2Bt, mask8, R, Cdim, Rp, Rp, Kp, 0);
    k_rowsum_recip_u8<<<g1(R), 256, 0, stream>>>(mask8, rowsc, R, Kp);
    k_u8f16<<<g1((long long)R * Kp), 256, 0, stream>>>(mask8, mask16, (long long)R * Kp);
  };
  auto attr_one = [&](int m, int gm, int R, int Kp, const _Float16* embT,
                      const float* Wp, const float* Bp, const float* Xm, const float* Vm,
                      const float* Wq, const float* Bq, const float* Qm) {
    dim3 g(cdiv(E_N, 32), cdiv(R, 64));
    k_gemm_f16<<<g, 128, 0, stream>>>(mask16, embT, h32, R, E_N, Kp, Kp, E_N, rowsc, nullptr, 0);
    k_attr_p<<<g1((long long)R * 3), 256, 0, stream>>>(h32, Wp + (size_t)gm * E_N * 3, Bp + gm * 3, p32, R);
    k_attr_H<<<g1((long long)R * E_N), 256, 0, stream>>>(p32, Xm + gm * 3, Vm + (size_t)gm * E_N * 3,
                                                         H3 + (size_t)m * R * E_N, R);
    k_zero<<<1, 32, 0, stream>>>(esc + m, 1);
    k_attr_score<<<g1(R), 256, 0, stream>>>(H3 + (size_t)m * R * E_N, Wq + (size_t)gm * E_N * 4,
                                            Bq + gm * 4, Qm + gm * 4, R, esc + m);
  };

  k_vecmat_col<<<g1(S_N), 256, 0, stream>>>(ua0, nullptr, vecS, U_N, S_N);
  k_matvec_row<<<g1(V_N), 256, 0, stream>>>(pa0, vecS, vecN, V_N, S_N);
  mask_rank1(U_N, V_N, V_P);
  attr_one(0, 0, U_N, V_P, prodT16, uWp, uBp, uX, uVv, uWq, uBq, uQ);
  k_vecmat_col<<<g1(S_N), 256, 0, stream>>>(pa0, dv_raw, vecS, V_N, S_N);
  k_matvec_row<<<g1(V_N), 256, 0, stream>>>(pa0, vecS, vecN, V_N, S_N);
  mask_rank1(U_N, V_N, V_P);
  attr_one(1, 1, U_N, V_P, prodT16, uWp, uBp, uX, uVv, uWq, uBq, uQ);
  mask_mp3(ua0_8, U_N, U_P, At8, V_N, V_P);
  attr_one(2, 2, U_N, V_P, prodT16, uWp, uBp, uX, uVv, uWq, uBq, uQ);
  k_attr_combine<<<g1((long long)U_N * E_N), 256, 0, stream>>>(H3, esc, 3, U_N, view2_u);
  k_vecmat_col<<<g1(S_N), 256, 0, stream>>>(pa0, nullptr, vecS, V_N, S_N);
  k_matvec_row<<<g1(U_N), 256, 0, stream>>>(ua0, vecS, vecN, U_N, S_N);
  mask_rank1(V_N, U_N, U_P);
  attr_one(0, 0, V_N, U_P, userT16, pWp, pBp, pX, pVv, pWq, pBq, pQ);
  k_vecmat_col<<<g1(S_N), 256, 0, stream>>>(ua0, du_raw, vecS, U_N, S_N);
  k_matvec_row<<<g1(U_N), 256, 0, stream>>>(ua0, vecS, vecN, U_N, S_N);
  mask_rank1(V_N, U_N, U_P);
  attr_one(1, 1, V_N, U_P, userT16, pWp, pBp, pX, pVv, pWq, pBq, pQ);
  mask_mp3(pa0_8, V_N, V_P, A8, U_N, U_P);
  attr_one(2, 2, V_N, U_P, userT16, pWp, pBp, pX, pVv, pWq, pBq, pQ);
  k_attr_combine<<<g1((long long)V_N * E_N), 256, 0, stream>>>(H3, esc, 3, V_N, view2_v);

  k_vecmat_col<<<g1(S_N), 256, 0, stream>>>(ua1, nullptr, vecS, U_N, S_N);
  k_matvec_row<<<g1(V_N), 256, 0, stream>>>(pa1, vecS, vecN, V_N, S_N);
  mask_rank1(U_N, V_N, V_P);
  attr_one(0, 3, U_N, V_P, prodT16, uWp, uBp, uX, uVv, uWq, uBq, uQ);
  k_vecmat_col<<<g1(S_N), 256, 0, stream>>>(pa1, dv_raw, vecS, V_N, S_N);
  k_matvec_row<<<g1(V_N), 256, 0, stream>>>(pa0, vecS, vecN, V_N, S_N);
  mask_rank1(U_N, V_N, V_P);
  attr_one(1, 4, U_N, V_P, prodT16, uWp, uBp, uX, uVv, uWq, uBq, uQ);
  mask_mp3(ua1_8, U_N, U_P, At8, V_N, V_P);
  attr_one(2, 5, U_N, V_P, prodT16, uWp, uBp, uX, uVv, uWq, uBq, uQ);
  k_attr_combine<<<g1((long long)U_N * E_N), 256, 0, stream>>>(H3, esc, 3, U_N, view3_u);
  k_vecmat_col<<<g1(S_N), 256, 0, stream>>>(pa1, nullptr, vecS, V_N, S_N);
  k_matvec_row<<<g1(U_N), 256, 0, stream>>>(ua1, vecS, vecN, U_N, S_N);
  mask_rank1(V_N, U_N, U_P);
  attr_one(0, 3, V_N, U_P, userT16, pWp, pBp, pX, pVv, pWq, pBq, pQ);
  k_vecmat_col<<<g1(S_N), 256, 0, stream>>>(ua1, du_raw, vecS, U_N, S_N);
  k_matvec_row<<<g1(U_N), 256, 0, stream>>>(ua1, vecS, vecN, U_N, S_N);
  mask_rank1(V_N, U_N, U_P);
  attr_one(1, 4, V_N, U_P, userT16, pWp, pBp, pX, pVv, pWq, pBq, pQ);
  mask_mp3(pa1_8, V_N, V_P, A8, U_N, U_P);
  attr_one(2, 5, V_N, U_P, userT16, pWp, pBp, pX, pVv, pWq, pBq, pQ);
  k_attr_combine<<<g1((long long)V_N * E_N), 256, 0, stream>>>(H3, esc, 3, V_N, view3_v);

  k_vecmat_col<<<g1(S_N), 256, 0, stream>>>(pa2, dv_raw, vecS, V_N, S_N);
  k_matvec_row<<<g1(V_N), 256, 0, stream>>>(pa2, vecS, vecN, V_N, S_N);
  mask_rank1(U_N, V_N, V_P);
  attr_one(0, 6, U_N, V_P, prodT16, uWp, uBp, uX, uVv, uWq, uBq, uQ);
  k_attr_combine<<<g1((long long)U_N * E_N), 256, 0, stream>>>(H3, esc, 1, U_N, view4_u);
  mask_mp3(pa2_8, V_N, V_P, A8, U_N, U_P);
  attr_one(0, 6, V_N, U_P, userT16, pWp, pBp, pX, pVv, pWq, pBq, pQ);
  k_attr_combine<<<g1((long long)V_N * E_N), 256, 0, stream>>>(H3, esc, 1, V_N, view4_v);

  k_fuse4<<<g1((long long)U_N * E_N), 256, 0, stream>>>(view1_u, view2_u, view3_u, view4_u, uvw,
                                                        (long long)U_N * E_N, out_u);
  k_fuse4<<<g1((long long)V_N * E_N), 256, 0, stream>>>(view1_v, view2_v, view3_v, view4_v, pvw,
                                                        (long long)V_N * E_N, out_v);
}
